// MultiheadLMAttentionWithCache_5188320493666
// MI455X (gfx1250) — matrix operand order repaired and checked by reading, not compiled, not run on hardware
//
#include <hip/hip_runtime.h>
#include <hip/hip_bf16.h>

// ---------------------------------------------------------------------------
// MultiheadLMAttentionWithCache for MI455X (gfx1250, wave32, WMMA f16->f32)
// ---------------------------------------------------------------------------
#define TGT    2048
#define BSZ    2
#define EMB    2048
#define NH     16
#define HD     128
#define CACHE  2048
#define SRC    4096           // CACHE + TGT
#define MROWS  4096           // TGT * BSZ

typedef __attribute__((ext_vector_type(16))) _Float16 v16h;
typedef __attribute__((ext_vector_type(8)))  _Float16 v8h;
typedef __attribute__((ext_vector_type(8)))  float    v8f;

static __device__ __forceinline__ v8f zero8() {
  v8f z = {0.f, 0.f, 0.f, 0.f, 0.f, 0.f, 0.f, 0.f};
  return z;
}

static __device__ __forceinline__ v8f wmma32(v16h a, v16h b, v8f c) {
  // v_wmma_f32_16x16x32_f16  D = A*B + C   (wave32, f32 accumulate)
  return __builtin_amdgcn_wmma_f32_16x16x32_f16(
      /*neg_a=*/false, a, /*neg_b=*/false, b,
      /*c_mod=*/(short)0, c, /*reuse_a=*/false, /*reuse_b=*/false);
}

// A-operand fragment (16x32 f16, MxK): lane<16 -> M=lane, K={0..7,16..23};
// lane>=16 -> M=lane-16, K={8..15,24..31}.  base points at (m=0,k=0), ld in halfs.
static __device__ __forceinline__ v16h loadA16(const _Float16* base, int ld, int lane) {
  const int ln = lane & 15, hf = lane >> 4;
  const _Float16* p = base + ln * ld + hf * 8;
  union { v16h v; v8h h[2]; } u;
  u.h[0] = *(const v8h*)(p);
  u.h[1] = *(const v8h*)(p + 16);
  return u.v;
}

// B-operand fragment (32x16 f16, KxN): lane<16 -> N=lane, K={0..7,16..23};
// lane>=16 -> N=lane-16, K={8..15,24..31}.  base points at (n=0,k=0), ld in halfs
// (rows indexed by N, K contiguous within a row).
static __device__ __forceinline__ v16h loadB16(const _Float16* base, int ld, int lane) {
  const int ln = lane & 15, hf = lane >> 4;
  const _Float16* p = base + ln * ld + hf * 8;
  union { v16h v; v8h h[2]; } u;
  u.h[0] = *(const v8h*)(p);
  u.h[1] = *(const v8h*)(p + 16);
  return u.v;
}

// ---------------------------------------------------------------------------
// fp32 cache copy: [32,2048,128] -> first 2048 rows of [32,4096,128]
// ---------------------------------------------------------------------------
__global__ __launch_bounds__(256) void copy_cache_kernel(const float* __restrict__ src,
                                                         float* __restrict__ dst) {
  const size_t i = (size_t)blockIdx.x * 256 + threadIdx.x;   // float4 index
  const float4 v = ((const float4*)src)[i];
  const size_t bh  = i >> 16;          // 2048*128/4 = 65536 float4 per head
  const size_t rem = i & 65535;
  ((float4*)dst)[bh * 131072 + rem] = v;   // 4096*128/4 = 131072 per head
}

// ---------------------------------------------------------------------------
// GEMM: C[M=4096, N=2048] = A[4096,2048] @ B[2048,2048]^T + bias   (torch Linear)
// fp32 inputs converted to f16 during LDS staging; WMMA f16 -> f32 accum.
// Block 256 thr = 8 waves; tile 128x128; waves 2(M) x 4(N); wave tile 64x32.
// EPI 0: q  -> f16 head layout [bH, T, D], * scale
// EPI 1: kv -> fp32 head layout [bH, SRC, D] at row CACHE+t
// EPI 2: out-> fp32 row-major [M, N]
// ---------------------------------------------------------------------------
template <int EPI>
__global__ __launch_bounds__(256) void gemm_kernel(const float* __restrict__ A,
                                                   const float* __restrict__ B,
                                                   const float* __restrict__ bias,
                                                   float* __restrict__ outF,
                                                   _Float16* __restrict__ outH,
                                                   float scale) {
  constexpr int LDT = 72;   // 64 + 8 halfs pad
  __shared__ __align__(16) _Float16 As[128 * LDT];
  __shared__ __align__(16) _Float16 Bs[128 * LDT];

  const int tid  = threadIdx.x;
  const int wave = tid >> 5, lane = tid & 31;
  const int ln = lane & 15, hf = lane >> 4;
  const int waveM = wave & 1, waveN = wave >> 1;
  const int mBase = blockIdx.y * 128;
  const int nBase = blockIdx.x * 128;

  v8f acc[4][2];
#pragma unroll
  for (int mi = 0; mi < 4; ++mi)
#pragma unroll
    for (int ni = 0; ni < 2; ++ni) acc[mi][ni] = zero8();

  for (int k0 = 0; k0 < EMB; k0 += 64) {
    // ---- stage 128x64 tiles of A and B (fp32 -> f16) -----------------------
#pragma unroll
    for (int i = 0; i < 8; ++i) {
      const int f4  = tid + i * 256;        // 0..2047
      const int row = f4 >> 4;              // 16 float4 per row of 64
      const int c4  = (f4 & 15) << 2;
      const float4 a = *(const float4*)(A + (size_t)(mBase + row) * EMB + k0 + c4);
      _Float16* da = &As[row * LDT + c4];
      da[0] = (_Float16)a.x; da[1] = (_Float16)a.y;
      da[2] = (_Float16)a.z; da[3] = (_Float16)a.w;
      const float4 b = *(const float4*)(B + (size_t)(nBase + row) * EMB + k0 + c4);
      _Float16* db = &Bs[row * LDT + c4];
      db[0] = (_Float16)b.x; db[1] = (_Float16)b.y;
      db[2] = (_Float16)b.z; db[3] = (_Float16)b.w;
    }
    __syncthreads();

    // ---- 2 k-chunks of 32, 8 WMMA each -------------------------------------
#pragma unroll
    for (int kc = 0; kc < 2; ++kc) {
      v16h af[4], bf[2];
#pragma unroll
      for (int mi = 0; mi < 4; ++mi)
        af[mi] = loadA16(&As[(waveM * 64 + mi * 16) * LDT + kc * 32], LDT, lane);
#pragma unroll
      for (int ni = 0; ni < 2; ++ni)
        bf[ni] = loadB16(&Bs[(waveN * 32 + ni * 16) * LDT + kc * 32], LDT, lane);
#pragma unroll
      for (int mi = 0; mi < 4; ++mi)
#pragma unroll
        for (int ni = 0; ni < 2; ++ni)
          acc[mi][ni] = wmma32(af[mi], bf[ni], acc[mi][ni]);
    }
    __syncthreads();
  }

  // ---- epilogue -------------------------------------------------------------
#pragma unroll
  for (int ni = 0; ni < 2; ++ni) {
    const int n  = nBase + waveN * 32 + ni * 16 + ln;
    const float bn = bias[n];
#pragma unroll
    for (int mi = 0; mi < 4; ++mi) {
#pragma unroll
      for (int r = 0; r < 8; ++r) {
        const int m = mBase + waveM * 64 + mi * 16 + r + 8 * hf;
        const float val = acc[mi][ni][r] + bn;
        if (EPI == 0) {
          const int t = m >> 1, b = m & 1, h = n >> 7, d = n & 127;
          outH[(((size_t)(b * NH + h)) * TGT + t) * HD + d] = (_Float16)(val * scale);
        } else if (EPI == 1) {
          const int t = m >> 1, b = m & 1, h = n >> 7, d = n & 127;
          outF[(((size_t)(b * NH + h)) * SRC + CACHE + t) * HD + d] = val;
        } else {
          outF[(size_t)m * EMB + n] = val;
        }
      }
    }
  }
}

// ---------------------------------------------------------------------------
// Flash attention per (bh, 64-query tile). 128 thr = 4 waves; each wave owns a
// 16-row Q tile in registers.  K staged row-major, V staged transposed, P
// bounced through LDS (C-layout -> A-layout reshape).  Causal mask applies
// only to keys >= CACHE; key loop terminates early past the causal frontier.
// ---------------------------------------------------------------------------
__global__ __launch_bounds__(128) void attn_kernel(const _Float16* __restrict__ qh,
                                                   const float* __restrict__ kf,
                                                   const float* __restrict__ vf,
                                                   float* __restrict__ attnOut) {
  constexpr int LDK = 136;  // 128 + 8
  constexpr int LDV = 72;   // 64 + 8
  constexpr int LDP = 72;
  __shared__ __align__(16) _Float16 kT[64 * LDK];        // [key][d]
  __shared__ __align__(16) _Float16 vT[128 * LDV];       // [d][key]
  __shared__ __align__(16) _Float16 pT[4 * 16 * LDP];    // per-wave [row][key]

  const int tid  = threadIdx.x;
  const int wave = tid >> 5, lane = tid & 31;
  const int ln = lane & 15, hf = lane >> 4;
  const int qt = blockIdx.x;       // 0..31 (64 queries each)
  const int bh = blockIdx.y;       // 0..31
  const int tBase = qt * 64 + wave * 16;

  // Q fragments (16x128 f16, pre-scaled) straight from global in A-layout
  const _Float16* qbase = qh + ((size_t)bh * TGT + tBase) * HD;
  v16h qfr[4];
#pragma unroll
  for (int kc = 0; kc < 4; ++kc) qfr[kc] = loadA16(qbase + kc * 32, HD, lane);

  v8f o[8];
#pragma unroll
  for (int dt = 0; dt < 8; ++dt) o[dt] = zero8();
  float mrow[8], lrow[8];
#pragma unroll
  for (int r = 0; r < 8; ++r) { mrow[r] = -1e30f; lrow[r] = 0.f; }

  const int jEnd = CACHE + (qt + 1) * 64;   // exclusive causal frontier
  for (int j0 = 0; j0 < jEnd; j0 += 64) {
    // ---- stage K[64][128] and V^T[128][64] (fp32 -> f16) -------------------
    const float* ksrc = kf + ((size_t)bh * SRC + j0) * HD;
    const float* vsrc = vf + ((size_t)bh * SRC + j0) * HD;
#pragma unroll
    for (int i = 0; i < 16; ++i) {
      const int f4  = tid + i * 128;        // 0..2047
      const int row = f4 >> 5;              // 32 float4 per 128-wide row
      const int c4  = (f4 & 31) << 2;
      const float4 a = *(const float4*)(ksrc + (size_t)row * HD + c4);
      _Float16* dk = &kT[row * LDK + c4];
      dk[0] = (_Float16)a.x; dk[1] = (_Float16)a.y;
      dk[2] = (_Float16)a.z; dk[3] = (_Float16)a.w;
      const float4 b = *(const float4*)(vsrc + (size_t)row * HD + c4);
      vT[(c4 + 0) * LDV + row] = (_Float16)b.x;
      vT[(c4 + 1) * LDV + row] = (_Float16)b.y;
      vT[(c4 + 2) * LDV + row] = (_Float16)b.z;
      vT[(c4 + 3) * LDV + row] = (_Float16)b.w;
    }
    __syncthreads();

    // ---- S = Q K^T for this wave's 16 rows x 64 keys -----------------------
    v8f s[4];
#pragma unroll
    for (int nt = 0; nt < 4; ++nt) {
      s[nt] = zero8();
#pragma unroll
      for (int kc = 0; kc < 4; ++kc) {
        v16h bf = loadB16(&kT[(nt * 16) * LDK + kc * 32], LDK, lane);
        s[nt] = wmma32(qfr[kc], bf, s[nt]);
      }
    }

    // ---- causal mask on key blocks past the cache --------------------------
    if (j0 + 63 > tBase + CACHE) {
#pragma unroll
      for (int nt = 0; nt < 4; ++nt) {
        const int j = j0 + nt * 16 + ln;
#pragma unroll
        for (int r = 0; r < 8; ++r) {
          const int t = tBase + r + 8 * hf;
          if (j - CACHE > t) s[nt][r] = -1e30f;
        }
      }
    }

    // ---- online softmax (row reductions across the 16-lane half) -----------
#pragma unroll
    for (int r = 0; r < 8; ++r) {
      float mx = s[0][r];
#pragma unroll
      for (int nt = 1; nt < 4; ++nt) mx = fmaxf(mx, s[nt][r]);
      mx = fmaxf(mx, __shfl_xor(mx, 8, 32));
      mx = fmaxf(mx, __shfl_xor(mx, 4, 32));
      mx = fmaxf(mx, __shfl_xor(mx, 2, 32));
      mx = fmaxf(mx, __shfl_xor(mx, 1, 32));
      const float mnew  = fmaxf(mrow[r], mx);
      const float corr  = __expf(mrow[r] - mnew);
      float sum = 0.f;
#pragma unroll
      for (int nt = 0; nt < 4; ++nt) {
        const float p = __expf(s[nt][r] - mnew);
        s[nt][r] = p;
        sum += p;
      }
      sum += __shfl_xor(sum, 8, 32);
      sum += __shfl_xor(sum, 4, 32);
      sum += __shfl_xor(sum, 2, 32);
      sum += __shfl_xor(sum, 1, 32);
      lrow[r] = lrow[r] * corr + sum;
      mrow[r] = mnew;
#pragma unroll
      for (int dt = 0; dt < 8; ++dt) o[dt][r] *= corr;
      // spill P row to per-wave LDS (C-layout -> row-major [16][64])
#pragma unroll
      for (int nt = 0; nt < 4; ++nt)
        pT[(wave * 16 + r + 8 * hf) * LDP + nt * 16 + ln] = (_Float16)s[nt][r];
    }

    // ---- O += P V  (P as A-operand from LDS, V^T as B-operand) -------------
#pragma unroll
    for (int kc2 = 0; kc2 < 2; ++kc2) {
      v16h pa = loadA16(&pT[(wave * 16) * LDP + kc2 * 32], LDP, lane);
#pragma unroll
      for (int dt = 0; dt < 8; ++dt) {
        v16h bv = loadB16(&vT[(dt * 16) * LDV + kc2 * 32], LDV, lane);
        o[dt] = wmma32(pa, bv, o[dt]);
      }
    }
    __syncthreads();
  }

  // ---- normalize and write [t*BSZ+b][h*128+d] fp32 ---------------------------
  const int b = bh >> 4, h = bh & 15;
#pragma unroll
  for (int r = 0; r < 8; ++r) {
    const float inv = 1.0f / lrow[r];
    const int t = tBase + r + 8 * hf;
    float* dst = attnOut + ((size_t)t * BSZ + b) * EMB + h * HD;
#pragma unroll
    for (int dt = 0; dt < 8; ++dt) dst[dt * 16 + ln] = o[dt][r] * inv;
  }
}

// ---------------------------------------------------------------------------
extern "C" void kernel_launch(void* const* d_in, const int* in_sizes, int n_in,
                              void* d_out, int out_size, void* d_ws, size_t ws_size,
                              hipStream_t stream) {
  (void)in_sizes; (void)n_in; (void)out_size; (void)ws_size;
  const float* x       = (const float*)d_in[0];
  const float* cache_k = (const float*)d_in[1];
  const float* cache_v = (const float*)d_in[2];
  const float* Wq = (const float*)d_in[3];
  const float* bq = (const float*)d_in[4];
  const float* Wk = (const float*)d_in[5];
  const float* bk = (const float*)d_in[6];
  const float* Wv = (const float*)d_in[7];
  const float* bv = (const float*)d_in[8];
  const float* Wo = (const float*)d_in[9];
  const float* bo = (const float*)d_in[10];

  float* out      = (float*)d_out;
  float* out_attn = out;                                   // [2048,2,2048]
  float* out_k    = out + (size_t)TGT * BSZ * EMB;         // [32,4096,128]
  float* out_v    = out_k + (size_t)BSZ * NH * SRC * HD;   // [32,4096,128]

  // workspace: q f16 (16 MB) + attn fp32 (32 MB) = 48 MB
  _Float16* q_h    = (_Float16*)d_ws;
  float*    attn_f = (float*)((char*)d_ws + (size_t)16 * 1024 * 1024);

  const float scaling = 0.08838834764831845f;   // 1/sqrt(128)

  // 1) fp32 cache -> K/V output slices (rows 0..2047)
  copy_cache_kernel<<<8192, 256, 0, stream>>>(cache_k, out_k);
  copy_cache_kernel<<<8192, 256, 0, stream>>>(cache_v, out_v);

  // 2) Q/K/V projections (WMMA). K/V write directly into d_out (rows 2048..4095).
  dim3 gg(16, 32);   // N/128 x M/128
  gemm_kernel<0><<<gg, 256, 0, stream>>>(x, Wq, bq, nullptr, q_h, scaling);
  gemm_kernel<1><<<gg, 256, 0, stream>>>(x, Wk, bk, out_k, nullptr, 1.f);
  gemm_kernel<1><<<gg, 256, 0, stream>>>(x, Wv, bv, out_v, nullptr, 1.f);

  // 3) flash attention over the concatenated K/V (reads d_out slices)
  dim3 ga(32, 32);   // q-tiles x (bsz*heads)
  attn_kernel<<<ga, 128, 0, stream>>>(q_h, out_k, out_v, attn_f);

  // 4) output projection -> d_out attn region
  gemm_kernel<2><<<gg, 256, 0, stream>>>(attn_f, Wo, bo, out_attn, nullptr, 1.f);
}
